// BasicAttention_9131100471744
// MI455X (gfx1250) — matrix operand order repaired and checked by reading, not compiled, not run on hardware
//
#include <hip/hip_runtime.h>

// MI455X / gfx1250, wave32. All GEMMs via v_wmma_f32_16x16x32_f16,
// 2x2 register blocking per wave, vectorized (b128) LDS softmax strip.

typedef __attribute__((ext_vector_type(16))) _Float16 v16h;
typedef __attribute__((ext_vector_type(8)))  _Float16 v8h;
typedef __attribute__((ext_vector_type(8)))  float    v8f;

constexpr int B_ = 8, C_ = 1024, L_ = 2048, A_ = 128;

// ---- WMMA fragment loaders (layouts per CDNA5 ISA §7.12.2) ----
// A-matrix 16x32 f16, source row-major [M,K]:
//   lane L: row M=(L&15); K = ((L>>4)*8) + {0..7, 16..23}  -> two 16B loads
__device__ __forceinline__ v16h frag_a(const _Float16* __restrict__ base,
                                       int lda, int m0, int k0) {
  int lane = threadIdx.x & 31;
  const _Float16* p = base + (size_t)(m0 + (lane & 15)) * lda + k0 + ((lane >> 4) << 3);
  v8h lo = *(const v8h*)(p);
  v8h hi = *(const v8h*)(p + 16);
  v16h r;
#pragma unroll
  for (int i = 0; i < 8; ++i) { r[i] = lo[i]; r[i + 8] = hi[i]; }
  return r;
}
// B-matrix 32x16 f16, source stored transposed [N,K]:
//   lane L: col N=(L&15); K = ((L>>4)*8) + {0..7, 16..23}  -> two 16B loads
__device__ __forceinline__ v16h frag_b(const _Float16* __restrict__ base,
                                       int ldb, int n0, int k0) {
  int lane = threadIdx.x & 31;
  const _Float16* p = base + (size_t)(n0 + (lane & 15)) * ldb + k0 + ((lane >> 4) << 3);
  v8h lo = *(const v8h*)(p);
  v8h hi = *(const v8h*)(p + 16);
  v16h r;
#pragma unroll
  for (int i = 0; i < 8; ++i) { r[i] = lo[i]; r[i + 8] = hi[i]; }
  return r;
}
__device__ __forceinline__ v8f wmma_f16(v16h a, v16h b, v8f c) {
  return __builtin_amdgcn_wmma_f32_16x16x32_f16(false, a, false, b, (short)0, c,
                                                false, false);
}
// C/D tile: lane holds column N=(lane&15), rows M = 8*(lane>>4)+r, r=0..7.

// ---- elementwise f32 -> f16 (weights) ----
__global__ void k_cvt(const float* __restrict__ in, _Float16* __restrict__ out, int n) {
  int i = blockIdx.x * 256 + threadIdx.x;
  if (i < n) out[i] = (_Float16)in[i];
}

// ---- x: cast to f16 (same layout) + tiled transpose to xT[l,c] ----
__global__ void k_xcast(const float* __restrict__ x, _Float16* __restrict__ x16,
                        _Float16* __restrict__ xT) {
  __shared__ _Float16 t[32][34];
  int b = blockIdx.z;
  int l0 = blockIdx.x * 32, c0 = blockIdx.y * 32;
  int tx = threadIdx.x, ty = threadIdx.y;
  const float* xb = x + (size_t)b * C_ * L_;
#pragma unroll
  for (int j = 0; j < 4; ++j) {
    int c = c0 + ty + 8 * j;
    _Float16 h = (_Float16)xb[(size_t)c * L_ + l0 + tx];
    x16[(size_t)b * C_ * L_ + (size_t)c * L_ + l0 + tx] = h;
    t[ty + 8 * j][tx] = h;
  }
  __syncthreads();
#pragma unroll
  for (int j = 0; j < 4; ++j) {
    int l = l0 + ty + 8 * j;
    xT[(size_t)b * L_ * C_ + (size_t)l * C_ + c0 + tx] = t[tx][ty + 8 * j];
  }
}

// ---- fused K/Q projection: keysT[l,a] = sum_c Wk[a,c]*x[c,l] + bk[a] ----
// M=a, N=l, K=c. Per wave: 1 a-tile x 2 l-tiles x {K,Q} = 4 accumulators;
// 4 WMMAs per 4 fragment loads.
__global__ __launch_bounds__(128) void k_kq(
    const _Float16* __restrict__ xT, const _Float16* __restrict__ wk,
    const _Float16* __restrict__ wq, const float* __restrict__ bk,
    const float* __restrict__ bq, _Float16* __restrict__ kT,
    _Float16* __restrict__ qT) {
  int b = blockIdx.y;
  int unit = blockIdx.x * 4 + (threadIdx.x >> 5);
  int mi  = unit & 7;        // a-tile (8)
  int ni2 = unit >> 3;       // l-tile pair (64)
  const _Float16* xTb = xT + (size_t)b * L_ * C_;
  v8f ck0 = {}, ck1 = {}, cq0 = {}, cq1 = {};
  for (int k = 0; k < C_; k += 32) {
    v16h ak = frag_a(wk, C_, mi * 16, k);
    v16h aq = frag_a(wq, C_, mi * 16, k);
    v16h b0 = frag_b(xTb, C_, ni2 * 32, k);
    v16h b1 = frag_b(xTb, C_, ni2 * 32 + 16, k);
    ck0 = wmma_f16(ak, b0, ck0);
    ck1 = wmma_f16(ak, b1, ck1);
    cq0 = wmma_f16(aq, b0, cq0);
    cq1 = wmma_f16(aq, b1, cq1);
  }
  int lane = threadIdx.x & 31;
  int n = lane & 15, hi = lane >> 4;
  int a0 = mi * 16 + hi * 8;
  float bkv[8], bqv[8];
#pragma unroll
  for (int r = 0; r < 8; ++r) { bkv[r] = bk[a0 + r]; bqv[r] = bq[a0 + r]; }
  v8h ok0, ok1, oq0, oq1;
#pragma unroll
  for (int r = 0; r < 8; ++r) {
    ok0[r] = (_Float16)(ck0[r] + bkv[r]);
    ok1[r] = (_Float16)(ck1[r] + bkv[r]);
    oq0[r] = (_Float16)(cq0[r] + bqv[r]);
    oq1[r] = (_Float16)(cq1[r] + bqv[r]);
  }
  size_t row0 = (size_t)b * L_ + ni2 * 32 + n;
  *(v8h*)(kT + row0 * A_ + a0)        = ok0;
  *(v8h*)(kT + (row0 + 16) * A_ + a0) = ok1;
  *(v8h*)(qT + row0 * A_ + a0)        = oq0;
  *(v8h*)(qT + (row0 + 16) * A_ + a0) = oq1;
}

// ---- logits + softmax fused. One wave per 16-wide l strip. ----
// valuesT[m,l] = sum_a queriesT[m,a]*keysT[l,a]; softmax over m (columns).
// LDS strip stored as 8-element chunks: element (m,n) at ((m>>3)*16+n)*8+(m&7)
// -> every LDS access is ds_*_b128, banks 4n mod 64 (conflict-free over n).
__global__ __launch_bounds__(32) void k_attn(const _Float16* __restrict__ kT,
                                             const _Float16* __restrict__ qT,
                                             _Float16* __restrict__ attnT) {
  __shared__ _Float16 slog[(L_ / 8) * 16 * 8];  // 64 KB
  int b = blockIdx.y;
  int l0 = blockIdx.x * 16;
  const _Float16* kTb = kT + (size_t)b * L_ * A_;
  const _Float16* qTb = qT + (size_t)b * L_ * A_;
  int lane = threadIdx.x & 31;
  int n = lane & 15, hi = lane >> 4;

  v16h bf[4];  // keys block 16x128 cached in registers, reused by 128 m-tiles
#pragma unroll
  for (int kk = 0; kk < 4; ++kk) bf[kk] = frag_b(kTb, A_, l0, kk * 32);

  float colmax = -1e30f;
  for (int mt = 0; mt < L_ / 16; ++mt) {
    v8f acc = {};
#pragma unroll
    for (int kk = 0; kk < 4; ++kk)
      acc = wmma_f16(frag_a(qTb, A_, mt * 16, kk * 32), bf[kk], acc);
    v8h sv;
#pragma unroll
    for (int r = 0; r < 8; ++r) {
      float s = acc[r] * (2.0f / (float)L_);  // /(L/2)
      colmax = fmaxf(colmax, s);
      sv[r] = (_Float16)s;
    }
    // lane's 8 rows are m = mt*16 + hi*8 + r -> chunk (mt*2+hi), column n
    *(v8h*)&slog[(((size_t)mt * 2 + hi) * 16 + n) * 8] = sv;
  }
  colmax = fmaxf(colmax, __shfl_xor(colmax, 16));  // lanes n, n+16 share a column
  __syncthreads();

  float sum = 0.0f;
  int cbeg = hi * (L_ / 16);  // lane's half-column: chunks [hi*128, hi*128+128)
  for (int ci = cbeg; ci < cbeg + L_ / 16; ++ci) {
    _Float16* p = &slog[((size_t)ci * 16 + n) * 8];
    v8h v = *(v8h*)p;
    v8h e;
#pragma unroll
    for (int r = 0; r < 8; ++r) {
      float ev = __expf((float)v[r] - colmax);
      sum += ev;
      e[r] = (_Float16)ev;
    }
    *(v8h*)p = e;
  }
  sum += __shfl_xor(sum, 16);
  float inv = 1.0f / sum;
  __syncthreads();

  _Float16* ab = attnT + (size_t)b * L_ * L_;
  for (int ci = cbeg; ci < cbeg + L_ / 16; ++ci) {
    v8h e = *(v8h*)&slog[((size_t)ci * 16 + n) * 8];
#pragma unroll
    for (int r = 0; r < 8; ++r)  // fixed m across lanes -> 32B-coalesced
      ab[(size_t)(ci * 8 + r) * L_ + l0 + n] = (_Float16)((float)e[r] * inv);
  }
}

// ---- out_mid^T[m,c] = sum_l x16[c,l] * attnT[m,l]; 2x2 tiles per wave ----
__global__ __launch_bounds__(128) void k_xattn(const _Float16* __restrict__ x16,
                                               const _Float16* __restrict__ attnT,
                                               _Float16* __restrict__ midT) {
  int b = blockIdx.y;
  int unit = blockIdx.x * 4 + (threadIdx.x >> 5);
  int mi2 = unit & 31;  // c-tile pair (32)
  int ni2 = unit >> 5;  // m-tile pair (64)
  const _Float16* xb = x16 + (size_t)b * C_ * L_;
  const _Float16* ab = attnT + (size_t)b * L_ * L_;
  v8f c00 = {}, c01 = {}, c10 = {}, c11 = {};
  for (int k = 0; k < L_; k += 32) {
    v16h a0 = frag_a(xb, L_, mi2 * 32, k);
    v16h a1 = frag_a(xb, L_, mi2 * 32 + 16, k);
    v16h b0 = frag_b(ab, L_, ni2 * 32, k);
    v16h b1 = frag_b(ab, L_, ni2 * 32 + 16, k);
    c00 = wmma_f16(a0, b0, c00);
    c01 = wmma_f16(a0, b1, c01);
    c10 = wmma_f16(a1, b0, c10);
    c11 = wmma_f16(a1, b1, c11);
  }
  int lane = threadIdx.x & 31;
  int n = lane & 15, hi = lane >> 4;
  int c0 = mi2 * 32 + hi * 8;
  size_t row = (size_t)b * L_ * C_ + (size_t)(ni2 * 32 + n) * C_;
  v8h o00, o01, o10, o11;
#pragma unroll
  for (int r = 0; r < 8; ++r) {
    o00[r] = (_Float16)c00[r]; o01[r] = (_Float16)c01[r];
    o10[r] = (_Float16)c10[r]; o11[r] = (_Float16)c11[r];
  }
  *(v8h*)(midT + row + c0)                 = o00;
  *(v8h*)(midT + row + 16 * C_ + c0)       = o01;
  *(v8h*)(midT + row + c0 + 16)            = o10;
  *(v8h*)(midT + row + 16 * C_ + c0 + 16)  = o11;
}

// ---- final[d,m] = sum_c Wp[d,c]*mid[c,m] + bp[d]; 2x2 tiles per wave ----
__global__ __launch_bounds__(128) void k_proj(const _Float16* __restrict__ wp,
                                              const _Float16* __restrict__ midT,
                                              const float* __restrict__ bp,
                                              float* __restrict__ out) {
  int b = blockIdx.y;
  int unit = blockIdx.x * 4 + (threadIdx.x >> 5);
  int mi2 = unit & 31;  // d-tile pair (32)
  int ni2 = unit >> 5;  // m-tile pair (64)
  const _Float16* mb = midT + (size_t)b * L_ * C_;
  v8f c00 = {}, c01 = {}, c10 = {}, c11 = {};
  for (int k = 0; k < C_; k += 32) {
    v16h a0 = frag_a(wp, C_, mi2 * 32, k);
    v16h a1 = frag_a(wp, C_, mi2 * 32 + 16, k);
    v16h b0 = frag_b(mb, C_, ni2 * 32, k);
    v16h b1 = frag_b(mb, C_, ni2 * 32 + 16, k);
    c00 = wmma_f16(a0, b0, c00);
    c01 = wmma_f16(a0, b1, c01);
    c10 = wmma_f16(a1, b0, c10);
    c11 = wmma_f16(a1, b1, c11);
  }
  int lane = threadIdx.x & 31;
  int n = lane & 15, hi = lane >> 4;
  int d0 = mi2 * 32 + hi * 8;
  int m0 = ni2 * 32 + n;
  float* ob = out + (size_t)b * C_ * L_;
#pragma unroll
  for (int r = 0; r < 8; ++r) {
    float bv0 = bp[d0 + r], bv1 = bp[d0 + 16 + r];
    ob[(size_t)(d0 + r) * L_ + m0]           = c00[r] + bv0;  // 64B-coalesced/half-wave
    ob[(size_t)(d0 + r) * L_ + m0 + 16]      = c01[r] + bv0;
    ob[(size_t)(d0 + 16 + r) * L_ + m0]      = c10[r] + bv1;
    ob[(size_t)(d0 + 16 + r) * L_ + m0 + 16] = c11[r] + bv1;
  }
}

extern "C" void kernel_launch(void* const* d_in, const int* in_sizes, int n_in,
                              void* d_out, int out_size, void* d_ws, size_t ws_size,
                              hipStream_t stream) {
  (void)in_sizes; (void)n_in; (void)out_size; (void)ws_size;
  const float* x  = (const float*)d_in[0];
  const float* Wk = (const float*)d_in[1];
  const float* bk = (const float*)d_in[2];
  const float* Wq = (const float*)d_in[3];
  const float* bq = (const float*)d_in[4];
  const float* Wp = (const float*)d_in[5];
  const float* bp = (const float*)d_in[6];
  float* out = (float*)d_out;

  char* ws = (char*)d_ws;
  size_t off = 0;
  auto alloc = [&](size_t elems) { _Float16* p = (_Float16*)(ws + off); off += elems * 2; return p; };
  _Float16* x16  = alloc((size_t)B_ * C_ * L_);  // 32 MB
  _Float16* xT   = alloc((size_t)B_ * L_ * C_);  // 32 MB
  _Float16* wk16 = alloc((size_t)A_ * C_);
  _Float16* wq16 = alloc((size_t)A_ * C_);
  _Float16* wp16 = alloc((size_t)C_ * C_);
  _Float16* kTh  = alloc((size_t)B_ * L_ * A_);  // 4 MB
  _Float16* qTh  = alloc((size_t)B_ * L_ * A_);  // 4 MB
  _Float16* attT = alloc((size_t)B_ * L_ * L_);  // 64 MB
  _Float16* midT = alloc((size_t)B_ * L_ * C_);  // 32 MB

  k_cvt<<<dim3((A_ * C_ + 255) / 256), dim3(256), 0, stream>>>(Wk, wk16, A_ * C_);
  k_cvt<<<dim3((A_ * C_ + 255) / 256), dim3(256), 0, stream>>>(Wq, wq16, A_ * C_);
  k_cvt<<<dim3((C_ * C_ + 255) / 256), dim3(256), 0, stream>>>(Wp, wp16, C_ * C_);
  k_xcast<<<dim3(L_ / 32, C_ / 32, B_), dim3(32, 8), 0, stream>>>(x, x16, xT);
  // K/Q: 8 a-tiles x 64 l-tile-pairs per batch, 4 waves per block
  k_kq<<<dim3((8 * 64) / 4, B_), dim3(128), 0, stream>>>(xT, wk16, wq16, bk, bq, kTh, qTh);
  // logits+softmax: one wave per 16-column strip
  k_attn<<<dim3(L_ / 16, B_), dim3(32), 0, stream>>>(kTh, qTh, attT);
  // x @ attn: 32x64 tile-pair units per batch
  k_xattn<<<dim3((32 * 64) / 4, B_), dim3(128), 0, stream>>>(x16, attT, midT);
  // output projection
  k_proj<<<dim3((32 * 64) / 4, B_), dim3(128), 0, stream>>>(wp16, midT, bp, out);
}
